// MultiheadSelfAttention_1279900254682
// MI455X (gfx1250) — matrix operand order repaired and checked by reading, not compiled, not run on hardware
//
#include <hip/hip_runtime.h>
#include <hip/hip_bf16.h>

#define EMBED 1024
#define HEADS 16
#define HDIM  64
#define NB    4
#define LSEQ  2048
#define SCALE 0.03125f   // 1/sqrt(1024)

typedef __attribute__((ext_vector_type(16))) __bf16 v16bf;
typedef __attribute__((ext_vector_type(8)))  __bf16 v8bf;
typedef __attribute__((ext_vector_type(8)))  float  v8f;

static __device__ __forceinline__ v8bf ld8(const __bf16* p) {
  return *reinterpret_cast<const v8bf*>(p);
}
static __device__ __forceinline__ v16bf cat16(v8bf a, v8bf b) {
  v16bf r;
#pragma unroll
  for (int i = 0; i < 8; ++i) { r[i] = a[i]; r[i + 8] = b[i]; }
  return r;
}
static __device__ __forceinline__ v8f zero8() {
  v8f r;
#pragma unroll
  for (int i = 0; i < 8; ++i) r[i] = 0.0f;
  return r;
}
static __device__ __forceinline__ v8f wmma_bf16(v16bf a, v16bf b, v8f c) {
  return __builtin_amdgcn_wmma_f32_16x16x32_bf16(false, a, false, b, (short)0, c,
                                                 false, false);
}

// ---------------------------------------------------------------------------
// Kernel 0: Wo (f32 [k][n]) -> WoT bf16 [n][k]  (contiguous writes)
// ---------------------------------------------------------------------------
__global__ __launch_bounds__(256) void cvt_wo_kernel(const float* __restrict__ Wo,
                                                     __bf16* __restrict__ WoT) {
  int idx = blockIdx.x * 256 + threadIdx.x;        // over EMBED*EMBED, [n][k]
  int n = idx / EMBED;
  int k = idx - n * EMBED;
  WoT[idx] = (__bf16)Wo[k * EMBED + n];
}

// ---------------------------------------------------------------------------
// Kernel 1: per-head QKV projection: X(16x64) @ W(64x64) -> bf16 [N][H][L][D]
// One wave per 16-row tile per head. 8 waves / block.
// ---------------------------------------------------------------------------
__global__ __launch_bounds__(256) void proj_kernel(
    const float* __restrict__ Vin, const float* __restrict__ Kin,
    const float* __restrict__ Qin, const float* __restrict__ Wv,
    const float* __restrict__ Wk, const float* __restrict__ Wq,
    __bf16* __restrict__ Vb, __bf16* __restrict__ Kb, __bf16* __restrict__ Qb) {
  const int wid  = blockIdx.x * 8 + (threadIdx.x >> 5);
  const int lane = threadIdx.x & 31;
  const bool lo  = lane < 16;
  const int lm   = lane & 15;

  const int TPL = LSEQ / 16;                 // 128 row tiles per (n,h)
  const int rt  = wid % TPL;
  const int h   = (wid / TPL) % HEADS;
  const int n   = wid / (TPL * HEADS);

  const float* Xs[3] = {Qin, Kin, Vin};
  const float* Ws[3] = {Wq, Wk, Wv};
  __bf16*      Os[3] = {Qb, Kb, Vb};

  for (int s = 0; s < 3; ++s) {
    const float* rowp =
        Xs[s] + (size_t)(n * LSEQ + rt * 16 + lm) * EMBED + h * HDIM;
    // A fragments (16 rows x K=64 in two K=32 chunks)
    v16bf a[2];
#pragma unroll
    for (int kc = 0; kc < 2; ++kc) {
      int k0 = kc * 32;
      int e0 = k0 + (lo ? 0 : 16);
      int e1 = k0 + 8 + (lo ? 0 : 16);
      v16bf t;
#pragma unroll
      for (int j = 0; j < 8; ++j) t[j] = (__bf16)rowp[e0 + j];
#pragma unroll
      for (int j = 0; j < 8; ++j) t[8 + j] = (__bf16)rowp[e1 + j];
      a[kc] = t;
    }
    const float* W = Ws[s];
    __bf16* obase = Os[s] + (size_t)((n * HEADS + h) * LSEQ + rt * 16) * HDIM;
#pragma unroll
    for (int i = 0; i < 4; ++i) {           // output N subtiles
      int col = i * 16 + lm;
      v8f c = zero8();
#pragma unroll
      for (int kc = 0; kc < 2; ++kc) {
        int kb = kc * 32 + (lo ? 0 : 16);
        v16bf b;
#pragma unroll
        for (int j = 0; j < 16; ++j) b[j] = (__bf16)W[(kb + j) * HDIM + col];
        c = wmma_bf16(a[kc], b, c);
      }
#pragma unroll
      for (int r = 0; r < 8; ++r) {
        int mrow = lo ? r : 8 + r;
        obase[mrow * HDIM + col] = (__bf16)c[r];
      }
    }
  }
}

// ---------------------------------------------------------------------------
// Kernel 2: flash attention. Block = 8 waves = 128 queries of one (n,h).
// LDS: K tile row-major [key][d], V tile transposed [d][key], per-wave P buf.
// ---------------------------------------------------------------------------
__global__ __launch_bounds__(256) void flash_kernel(
    const __bf16* __restrict__ Qb, const __bf16* __restrict__ Kb,
    const __bf16* __restrict__ Vb, const int* __restrict__ mask,
    __bf16* __restrict__ Ob) {
  __shared__ __bf16 Kt[64 * 64];        // 8 KB
  __shared__ __bf16 VtT[64 * 64];       // 8 KB (transposed)
  __shared__ __bf16 Pbuf[8 * 16 * 64];  // 16 KB (per-wave 16x64)

  const int tid  = threadIdx.x;
  const int wave = tid >> 5;
  const int lane = tid & 31;
  const bool lo  = lane < 16;
  const int lm   = lane & 15;

  const int QBLK = LSEQ / 128;          // 16
  const int qblk = blockIdx.x % QBLK;
  const int h    = (blockIdx.x / QBLK) % HEADS;
  const int n    = blockIdx.x / (QBLK * HEADS);
  const int qbase = qblk * 128 + wave * 16;

  const __bf16* qrow =
      Qb + (size_t)((n * HEADS + h) * LSEQ + qbase + lm) * HDIM;
  const int aoff = lo ? 0 : 8;
  v16bf qa0 = cat16(ld8(qrow + 0 + aoff), ld8(qrow + 16 + aoff));
  v16bf qa1 = cat16(ld8(qrow + 32 + aoff), ld8(qrow + 48 + aoff));

  v8f o[4];
#pragma unroll
  for (int i = 0; i < 4; ++i) o[i] = zero8();
  float mst[8], lst[8];
#pragma unroll
  for (int r = 0; r < 8; ++r) { mst[r] = -3.0e38f; lst[r] = 0.0f; }

  const __bf16* Kbase = Kb + (size_t)((n * HEADS + h) * LSEQ) * HDIM;
  const __bf16* Vbase = Vb + (size_t)((n * HEADS + h) * LSEQ) * HDIM;
  const int boff = lo ? 0 : 8;

  for (int t = 0; t < LSEQ / 64; ++t) {
    const int kbase = t * 64;
    // --- cooperative stage: K tile straight copy (8KB), V tile transposed ---
    {
      const uint4* ks4 = (const uint4*)(Kbase + (size_t)kbase * HDIM);
      uint4* kt4 = (uint4*)Kt;
      kt4[tid] = ks4[tid];
      kt4[tid + 256] = ks4[tid + 256];

      int key = tid & 63;
      int dc  = (tid >> 6) << 4;
      const __bf16* vp = Vbase + (size_t)(kbase + key) * HDIM + dc;
      v8bf va = ld8(vp), vb2 = ld8(vp + 8);
#pragma unroll
      for (int j = 0; j < 8; ++j) VtT[(dc + j) * 64 + key] = va[j];
#pragma unroll
      for (int j = 0; j < 8; ++j) VtT[(dc + 8 + j) * 64 + key] = vb2[j];
    }
    if (t + 1 < LSEQ / 64) {  // gfx1250 global_prefetch_b8 of next tiles
      __builtin_prefetch(Kbase + (size_t)(kbase + 64) * HDIM + tid * 16, 0, 1);
      __builtin_prefetch(Vbase + (size_t)(kbase + 64) * HDIM + tid * 16, 0, 1);
    }
    __syncthreads();

    // --- S = Q @ K^T ---
    v8f s[4];
#pragma unroll
    for (int i = 0; i < 4; ++i) {
      const __bf16* kp = Kt + (i * 16 + lm) * 64 + boff;
      v16bf b0 = cat16(ld8(kp + 0), ld8(kp + 16));
      v16bf b1 = cat16(ld8(kp + 32), ld8(kp + 48));
      v8f c = zero8();
      c = wmma_bf16(qa0, b0, c);
      c = wmma_bf16(qa1, b1, c);
      s[i] = c;
    }
    // --- scale + mask ---
#pragma unroll
    for (int i = 0; i < 4; ++i) {
      int keyg = kbase + i * 16 + lm;
#pragma unroll
      for (int r = 0; r < 8; ++r) {
        int qg = qbase + (lo ? r : 8 + r);
        float sv = s[i][r] * SCALE;
        if (mask[qg * LSEQ + keyg] == 0) sv = -1.0e20f;
        s[i][r] = sv;
      }
    }
    // --- online softmax row stats (rows live in lane-halves) ---
    float alpha[8];
#pragma unroll
    for (int r = 0; r < 8; ++r) {
      float rm = fmaxf(fmaxf(s[0][r], s[1][r]), fmaxf(s[2][r], s[3][r]));
      rm = fmaxf(rm, __shfl_xor(rm, 1, 16));
      rm = fmaxf(rm, __shfl_xor(rm, 2, 16));
      rm = fmaxf(rm, __shfl_xor(rm, 4, 16));
      rm = fmaxf(rm, __shfl_xor(rm, 8, 16));
      float mn = fmaxf(mst[r], rm);
      alpha[r] = __expf(mst[r] - mn);
      mst[r] = mn;
    }
    // --- P = exp(S - m), stash to LDS (C-layout -> A-layout bounce) ---
    float rs[8];
#pragma unroll
    for (int r = 0; r < 8; ++r) rs[r] = 0.0f;
    __bf16* Pw = Pbuf + wave * (16 * 64);
#pragma unroll
    for (int i = 0; i < 4; ++i) {
#pragma unroll
      for (int r = 0; r < 8; ++r) {
        float p = __expf(s[i][r] - mst[r]);
        rs[r] += p;
        int m = lo ? r : 8 + r;
        Pw[m * 64 + i * 16 + lm] = (__bf16)p;
      }
    }
#pragma unroll
    for (int r = 0; r < 8; ++r) {
      float v = rs[r];
      v += __shfl_xor(v, 1, 16);
      v += __shfl_xor(v, 2, 16);
      v += __shfl_xor(v, 4, 16);
      v += __shfl_xor(v, 8, 16);
      lst[r] = lst[r] * alpha[r] + v;
    }
#pragma unroll
    for (int i = 0; i < 4; ++i)
#pragma unroll
      for (int r = 0; r < 8; ++r) o[i][r] *= alpha[r];

    // --- O += P @ V ---
    const __bf16* pr = Pw + lm * 64;
    v16bf pa0 = cat16(ld8(pr + 0 + aoff), ld8(pr + 16 + aoff));
    v16bf pa1 = cat16(ld8(pr + 32 + aoff), ld8(pr + 48 + aoff));
#pragma unroll
    for (int i = 0; i < 4; ++i) {
      const __bf16* vtp = VtT + (i * 16 + lm) * 64 + boff;
      v16bf vb0 = cat16(ld8(vtp + 0), ld8(vtp + 16));
      v16bf vb1 = cat16(ld8(vtp + 32), ld8(vtp + 48));
      o[i] = wmma_bf16(pa0, vb0, o[i]);
      o[i] = wmma_bf16(pa1, vb1, o[i]);
    }
    __syncthreads();
  }

  // --- normalize and write O into concat layout [N][L][EMBED] (bf16) ---
#pragma unroll
  for (int i = 0; i < 4; ++i) {
    int col = h * HDIM + i * 16 + lm;
#pragma unroll
    for (int r = 0; r < 8; ++r) {
      int qg = qbase + (lo ? r : 8 + r);
      float val = o[i][r] / lst[r];
      Ob[(size_t)(n * LSEQ + qg) * EMBED + col] = (__bf16)val;
    }
  }
}

// ---------------------------------------------------------------------------
// Kernel 3: out[8192,1024] = Obf @ Wo + bo  (bf16 WMMA, f32 out)
// Block tile: M=128 (8 waves x 16) by N=64.
// ---------------------------------------------------------------------------
__global__ __launch_bounds__(256) void outproj_kernel(
    const __bf16* __restrict__ Ob, const __bf16* __restrict__ WoT,
    const float* __restrict__ bo, float* __restrict__ out) {
  const int wave = threadIdx.x >> 5;
  const int lane = threadIdx.x & 31;
  const bool lo  = lane < 16;
  const int lm   = lane & 15;
  const int mt   = blockIdx.x / 16;
  const int nt   = blockIdx.x % 16;
  const int Rb   = mt * 128 + wave * 16;
  const int Cb   = nt * 64;
  const int aoff = lo ? 0 : 16;
  const int boff = lo ? 0 : 16;

  v8f acc[4];
#pragma unroll
  for (int i = 0; i < 4; ++i) acc[i] = zero8();

  for (int kc = 0; kc < EMBED / 32; ++kc) {
    const int k0 = kc * 32;
    const __bf16* arow = Ob + (size_t)(Rb + lm) * EMBED + k0 + aoff;
    v16bf a = cat16(ld8(arow), ld8(arow + 8));
#pragma unroll
    for (int i = 0; i < 4; ++i) {
      const __bf16* bp = WoT + (size_t)(Cb + i * 16 + lm) * EMBED + k0 + boff;
      v16bf b = cat16(ld8(bp), ld8(bp + 8));
      acc[i] = wmma_bf16(a, b, acc[i]);
    }
  }
#pragma unroll
  for (int i = 0; i < 4; ++i) {
    int col = Cb + i * 16 + lm;
    float bb = bo[col];
#pragma unroll
    for (int r = 0; r < 8; ++r) {
      int row = Rb + (lo ? r : 8 + r);
      out[(size_t)row * EMBED + col] = acc[i][r] + bb;
    }
  }
}

// ---------------------------------------------------------------------------
extern "C" void kernel_launch(void* const* d_in, const int* in_sizes, int n_in,
                              void* d_out, int out_size, void* d_ws,
                              size_t ws_size, hipStream_t stream) {
  const float* values = (const float*)d_in[0];
  const float* keys   = (const float*)d_in[1];
  const float* query  = (const float*)d_in[2];
  const int*   mask   = (const int*)d_in[3];
  const float* Wv     = (const float*)d_in[4];
  const float* Wk     = (const float*)d_in[5];
  const float* Wq     = (const float*)d_in[6];
  const float* Wo     = (const float*)d_in[7];
  const float* bo     = (const float*)d_in[8];
  float* out = (float*)d_out;

  const size_t SZ = (size_t)NB * HEADS * LSEQ * HDIM;  // 8,388,608 elems
  __bf16* Qbf = (__bf16*)d_ws;
  __bf16* Kbf = Qbf + SZ;
  __bf16* Vbf = Kbf + SZ;
  __bf16* Obf = Vbf + SZ;                              // [N][L][EMBED]
  __bf16* WoT = Obf + (size_t)NB * LSEQ * EMBED;       // [n][k]

  cvt_wo_kernel<<<(EMBED * EMBED) / 256, 256, 0, stream>>>(Wo, WoT);
  proj_kernel<<<(NB * HEADS * (LSEQ / 16)) / 8, 256, 0, stream>>>(
      values, keys, query, Wv, Wk, Wq, Vbf, Kbf, Qbf);
  flash_kernel<<<NB * HEADS * (LSEQ / 128), 256, 0, stream>>>(Qbf, Kbf, Vbf,
                                                              mask, Obf);
  outproj_kernel<<<(NB * LSEQ / 128) * (EMBED / 64), 256, 0, stream>>>(
      Obf, WoT, bo, out);
}
